// BiLSTM_77996606096016
// MI455X (gfx1250) — matrix operand order repaired and checked by reading, not compiled, not run on hardware
//
#include <hip/hip_runtime.h>
#include <math.h>

typedef __bf16 bf16_t;
typedef __attribute__((ext_vector_type(16))) __bf16 v16bf;
typedef __attribute__((ext_vector_type(8)))  __bf16 v8bf;
typedef __attribute__((ext_vector_type(8)))  float  v8f;

#define T_SEQ 2048
#define H_DIM 1024
#define IN0   1088
#define IN1   2048
#define G4    4096   // 4*H (gate rows per direction)
#define NPROJ 8192   // 2 dirs * 4H
#define TAGS  50

// ---------------- CDNA5 async global->LDS helpers ----------------
// GLOBAL_LOAD_ASYNC_TO_LDS_B128 (GV mode): per-lane 16B memory->LDS copy that
// bypasses VGPRs; tracked by ASYNCcnt (CDNA5 ISA).

__device__ __forceinline__ void async_b128(unsigned lds_off, const void* gaddr){
  asm volatile("global_load_async_to_lds_b128 %0, %1, off"
               :: "v"(lds_off), "v"(gaddr) : "memory");
}
__device__ __forceinline__ void wait_async0(){
  asm volatile("s_wait_asynccnt 0x0" ::: "memory");
}

// ---------------- elementwise / prep kernels ----------------

__global__ void k_cvt_bf16(const float* __restrict__ src, bf16_t* __restrict__ dst, size_t n){
  size_t i = (size_t)blockIdx.x * blockDim.x + threadIdx.x;
  size_t stride = (size_t)gridDim.x * blockDim.x;
  for (; i < n; i += stride) dst[i] = (bf16_t)src[i];
}

__global__ void k_bias_add(const float* __restrict__ a, const float* __restrict__ b,
                           float* __restrict__ dst, int n){
  int i = blockIdx.x * blockDim.x + threadIdx.x;
  if (i < n) dst[i] = a[i] + b[i];
}

__global__ void k_pad_wout(const float* __restrict__ w, bf16_t* __restrict__ dst){
  // dst: [256][2048] bf16, rows >= TAGS zero-padded
  int i = blockIdx.x * blockDim.x + threadIdx.x;   // 256*2048 threads total
  int row = i >> 11, col = i & 2047;
  dst[i] = (row < TAGS) ? (bf16_t)w[row * 2048 + col] : (bf16_t)0.0f;
}

__global__ void k_embed(const float* __restrict__ sent, const int* __restrict__ tags,
                        const float* __restrict__ emb, bf16_t* __restrict__ x0){
  int t = blockIdx.x;
  int tag = tags[t];
  for (int c = threadIdx.x; c < IN0; c += blockDim.x){
    float v = (c < 1024) ? sent[(size_t)t * 1024 + c] : emb[tag * 64 + (c - 1024)];
    x0[(size_t)t * IN0 + c] = (bf16_t)v;
  }
}

__global__ void k_init_state(bf16_t* __restrict__ hbuf, int* __restrict__ bar){
  // hbuf: 2 layers * 2 parities * 2 dirs * 1024 = 8192 bf16
  int i = threadIdx.x;
  for (int j = i; j < 8192; j += 256) hbuf[j] = (bf16_t)0.0f;
  if (i < 8) bar[i] = 0;
}

// ---------------- WMMA bf16 GEMM:  C[M,N] = A[M,K] * B[N,K]^T + bias ----------------
// Block tile 64(M) x 256(N), K panel 32, double-buffered LDS filled by the
// async global->LDS engine. 8 waves, each wave owns a 32x64 tile
// (2x4 of v_wmma_f32_16x16x32_bf16). Fragments gathered per the CDNA5
// 16-bit A(16x32)/B(32x16) VGPR layouts (CDNA5 ISA §7.12.2).

#define BM 64
#define BN 256
#define BK 32
#define LDT 40   // padded LDS row stride in bf16 (80B rows keep 16B alignment)

__global__ __launch_bounds__(256)
void k_gemm_bf16(const bf16_t* __restrict__ A, const bf16_t* __restrict__ B,
                 const float* __restrict__ bias, float* __restrict__ C,
                 int K, int ldc, int nvalid)
{
  __shared__ __align__(16) bf16_t As[2][BM][LDT];
  __shared__ __align__(16) bf16_t Bs[2][BN][LDT];

  const int tid  = threadIdx.x;
  const int lane = tid & 31;
  const int wid  = tid >> 5;     // 0..7
  const int wm   = wid >> 2;     // 0..1 (two 32-row wave strips)
  const int wn   = wid & 3;      // 0..3 (four 64-col wave strips)
  const int mr   = lane & 15;    // row (A) / col (B) within 16
  const int kh   = lane >> 4;    // K-half selector

  const size_t mbase = (size_t)blockIdx.y * BM;
  const size_t nbase = (size_t)blockIdx.x * BN;

  // async fill mapping: A = 64 rows x 4 x 16B chunks (1 chunk/thread),
  //                     B = 256 rows x 4 x 16B chunks (1 row/thread)
  const int arow = tid >> 2, ac8 = (tid & 3) * 8;

  const bf16_t* Ag = &A[(mbase + arow) * (size_t)K + ac8];
  const bf16_t* Bg = &B[(nbase + tid) * (size_t)K];

  auto prefetch = [&](int buf, int kb){
    async_b128((unsigned)(size_t)&As[buf][arow][ac8], Ag + kb);
    const unsigned boff = (unsigned)(size_t)&Bs[buf][tid][0];
    const bf16_t* bp = Bg + kb;
    async_b128(boff,      bp);
    async_b128(boff + 16, bp + 8);
    async_b128(boff + 32, bp + 16);
    async_b128(boff + 48, bp + 24);
  };

  v8f acc[2][4] = {};
  const int nk = K / BK;
  prefetch(0, 0);

  for (int ki = 0; ki < nk; ++ki){
    wait_async0();        // my async writes of buf (ki&1) are in LDS
    __syncthreads();      // ... and everyone's; everyone done reading (ki+1)&1
    if (ki + 1 < nk) prefetch((ki + 1) & 1, (ki + 1) * BK);

    const int buf = ki & 1;
    v16bf a[2], b[4];
    #pragma unroll
    for (int mt = 0; mt < 2; ++mt){
      const int row = wm * 32 + mt * 16 + mr;
      #pragma unroll
      for (int j = 0; j < 8; ++j){
        // A 16x32 bf16: K = (j>=4)*16 + kh*8 + (j%4)*2
        const int k = ((j & 4) << 2) + kh * 8 + ((j & 3) << 1);
        a[mt][2*j]   = As[buf][row][k];
        a[mt][2*j+1] = As[buf][row][k + 1];
      }
    }
    #pragma unroll
    for (int nt = 0; nt < 4; ++nt){
      const int row = wn * 64 + nt * 16 + mr;
      #pragma unroll
      for (int j = 0; j < 8; ++j){
        // B 32x16 bf16: K = (j>=4)*16 + kh*8 + (j%4)*2
        const int k = ((j & 4) << 2) + kh * 8 + ((j & 3) << 1);
        b[nt][2*j]   = Bs[buf][row][k];
        b[nt][2*j+1] = Bs[buf][row][k + 1];
      }
    }
    #pragma unroll
    for (int mt = 0; mt < 2; ++mt)
      #pragma unroll
      for (int nt = 0; nt < 4; ++nt)
        acc[mt][nt] = __builtin_amdgcn_wmma_f32_16x16x32_bf16(
            false, a[mt], false, b[nt], (short)0, acc[mt][nt], false, false);
  }

  // C/D layout: VGPR e, lane l -> row = (l>>4)*8 + e, col = l&15
  #pragma unroll
  for (int mt = 0; mt < 2; ++mt){
    #pragma unroll
    for (int nt = 0; nt < 4; ++nt){
      const size_t r0 = mbase + wm * 32 + mt * 16 + kh * 8;
      const int    cc = (int)nbase + wn * 64 + nt * 16 + mr;
      if (cc < nvalid){
        const float bv = bias ? bias[cc] : 0.0f;
        #pragma unroll
        for (int e = 0; e < 8; ++e)
          C[(r0 + e) * (size_t)ldc + cc] = acc[mt][nt][e] + bv;
      }
    }
  }
}

// ---------------- persistent bidirectional LSTM recurrence ----------------
// grid = (32 blocks per direction, 2 directions). Block b owns hidden units
// [b*32, b*32+32): computes gate rows {g*1024 + j} for g=i,f,g,o so the cell
// update is block-local. Global spin barrier (monotonic counter) + parity
// double-buffered h between steps. W_hh (bf16, 33MB total) stays L2-resident.

__device__ __forceinline__ float sigmoidf_(float x){ return 1.0f / (1.0f + expf(-x)); }

__global__ __launch_bounds__(256)
void k_recur(const float* __restrict__ xp,    // [T, 8192] precomputed x-projections
             const bf16_t* __restrict__ whh,  // [2, 4096, 1024] bf16
             bf16_t* __restrict__ hout,       // [T, 2048] (col offset dir*1024)
             bf16_t* __restrict__ hbuf,       // [2 parity][2 dir][1024] bf16
             int* __restrict__ bar,           // [2] monotonic arrive counters
             int nblk)
{
  __shared__ float hprev[H_DIM];
  __shared__ float gacc[128];
  __shared__ float cstate[32];

  const int tid  = threadIdx.x;
  const int dir  = blockIdx.y;
  const int j0   = blockIdx.x * 32;

  const int rid  = tid >> 1;      // 0..127: which gate row
  const int gate = rid >> 5;      // 0..3  (i,f,g,o)
  const int jj   = rid & 31;
  const int half = tid & 1;       // K-half of the dot product

  const v8bf* wv = (const v8bf*)(whh + ((size_t)dir * G4 + gate * H_DIM + j0 + jj) * H_DIM
                                 + half * 512);
  const float* xcol = xp + (size_t)dir * G4 + gate * H_DIM + j0 + jj;

  if (tid < 32) cstate[tid] = 0.0f;

  for (int s = 0; s < T_SEQ; ++s){
    const int t = dir ? (T_SEQ - 1 - s) : s;

    __syncthreads();
    {
      const bf16_t* hsrc = hbuf + (size_t)(s & 1) * 2048 + dir * H_DIM;
      for (int i = tid; i < H_DIM; i += 256) hprev[i] = (float)hsrc[i];
    }
    __syncthreads();

    float acc = 0.0f;
    const float* hp = hprev + half * 512;
    #pragma unroll 4
    for (int k8 = 0; k8 < 64; ++k8){
      v8bf w = wv[k8];
      const float* hpp = hp + k8 * 8;
      #pragma unroll
      for (int e = 0; e < 8; ++e) acc += (float)w[e] * hpp[e];
    }
    acc += __shfl_xor(acc, 1);
    if (half == 0) gacc[rid] = acc + xcol[(size_t)t * NPROJ];
    __syncthreads();

    if (tid < 32){
      const int j = tid;
      const float ig = sigmoidf_(gacc[j]);
      const float fg = sigmoidf_(gacc[32 + j]);
      const float gg = tanhf(gacc[64 + j]);
      const float og = sigmoidf_(gacc[96 + j]);
      const float c  = fg * cstate[j] + ig * gg;
      cstate[j] = c;
      const float h  = og * tanhf(c);
      hout[(size_t)t * IN1 + dir * H_DIM + j0 + j] = (bf16_t)h;
      hbuf[(size_t)((s + 1) & 1) * 2048 + dir * H_DIM + j0 + j] = (bf16_t)h;
      __threadfence();
    }
    __syncthreads();
    if (tid == 0){
      atomicAdd(&bar[dir], 1);
      const int target = nblk * (s + 1);
      while (atomicAdd(&bar[dir], 0) < target) __builtin_amdgcn_s_sleep(2);
    }
    __syncthreads();
  }
}

// ---------------- launcher ----------------

extern "C" void kernel_launch(void* const* d_in, const int* in_sizes, int n_in,
                              void* d_out, int out_size, void* d_ws, size_t ws_size,
                              hipStream_t stream) {
  const float* sentence  = (const float*)d_in[0];
  const int*   sp_tags   = (const int*)  d_in[1];
  const float* emb_sp    = (const float*)d_in[2];
  const float* w_ih0     = (const float*)d_in[3];
  const float* w_hh0     = (const float*)d_in[4];
  const float* b_ih0     = (const float*)d_in[5];
  const float* b_hh0     = (const float*)d_in[6];
  const float* w_ih1     = (const float*)d_in[7];
  const float* w_hh1     = (const float*)d_in[8];
  const float* b_ih1     = (const float*)d_in[9];
  const float* b_hh1     = (const float*)d_in[10];
  const float* w_out     = (const float*)d_in[11];
  const float* b_out     = (const float*)d_in[12];
  float* out = (float*)d_out;

  // bump allocator on d_ws
  char* p = (char*)d_ws;
  auto alloc = [&](size_t bytes) -> void* {
    void* r = (void*)p;
    p += (bytes + 255) & ~(size_t)255;
    return r;
  };
  bf16_t* X0bf    = (bf16_t*)alloc((size_t)T_SEQ * IN0 * 2);
  bf16_t* Wih0bf  = (bf16_t*)alloc((size_t)2 * G4 * IN0 * 2);
  bf16_t* Whh0bf  = (bf16_t*)alloc((size_t)2 * G4 * H_DIM * 2);
  bf16_t* Wih1bf  = (bf16_t*)alloc((size_t)2 * G4 * IN1 * 2);
  bf16_t* Whh1bf  = (bf16_t*)alloc((size_t)2 * G4 * H_DIM * 2);
  bf16_t* WoutPad = (bf16_t*)alloc((size_t)256 * IN1 * 2);
  float*  bias0   = (float*) alloc((size_t)NPROJ * 4);
  float*  bias1   = (float*) alloc((size_t)NPROJ * 4);
  float*  XP0     = (float*) alloc((size_t)T_SEQ * NPROJ * 4);
  float*  XP1     = (float*) alloc((size_t)T_SEQ * NPROJ * 4);
  bf16_t* H1      = (bf16_t*)alloc((size_t)T_SEQ * IN1 * 2);
  bf16_t* H2      = (bf16_t*)alloc((size_t)T_SEQ * IN1 * 2);
  bf16_t* hbuf    = (bf16_t*)alloc((size_t)8192 * 2);  // 2 layers x 2 parity x 2 dir x 1024
  int*    bar     = (int*)   alloc((size_t)8 * 4);

  // 1) weight / input prep
  k_cvt_bf16<<<2048, 256, 0, stream>>>(w_ih0, Wih0bf, (size_t)2 * G4 * IN0);
  k_cvt_bf16<<<2048, 256, 0, stream>>>(w_hh0, Whh0bf, (size_t)2 * G4 * H_DIM);
  k_cvt_bf16<<<2048, 256, 0, stream>>>(w_ih1, Wih1bf, (size_t)2 * G4 * IN1);
  k_cvt_bf16<<<2048, 256, 0, stream>>>(w_hh1, Whh1bf, (size_t)2 * G4 * H_DIM);
  k_bias_add<<<NPROJ / 256, 256, 0, stream>>>(b_ih0, b_hh0, bias0, NPROJ);
  k_bias_add<<<NPROJ / 256, 256, 0, stream>>>(b_ih1, b_hh1, bias1, NPROJ);
  k_pad_wout<<<(256 * IN1) / 256, 256, 0, stream>>>(w_out, WoutPad);
  k_embed<<<T_SEQ, 256, 0, stream>>>(sentence, sp_tags, emb_sp, X0bf);
  k_init_state<<<1, 256, 0, stream>>>(hbuf, bar);

  // 2) layer-0 input projection (both directions fused along N): [2048,1088]x[8192,1088]^T
  k_gemm_bf16<<<dim3(NPROJ / BN, T_SEQ / BM), 256, 0, stream>>>(
      X0bf, Wih0bf, bias0, XP0, IN0, NPROJ, NPROJ);

  // 3) layer-0 recurrence (fwd+bwd concurrently) -> H1 [T, 2048]
  k_recur<<<dim3(H_DIM / 32, 2), 256, 0, stream>>>(
      XP0, Whh0bf, H1, hbuf + 0, bar + 0, H_DIM / 32);

  // 4) layer-1 input projection: [2048,2048]x[8192,2048]^T
  k_gemm_bf16<<<dim3(NPROJ / BN, T_SEQ / BM), 256, 0, stream>>>(
      H1, Wih1bf, bias1, XP1, IN1, NPROJ, NPROJ);

  // 5) layer-1 recurrence -> H2 [T, 2048]
  k_recur<<<dim3(H_DIM / 32, 2), 256, 0, stream>>>(
      XP1, Whh1bf, H2, hbuf + 4096, bar + 2, H_DIM / 32);

  // 6) output projection: [2048,2048]x[256(pad 50),2048]^T -> d_out [2048,50]
  k_gemm_bf16<<<dim3(1, T_SEQ / BM), 256, 0, stream>>>(
      H2, WoutPad, b_out, out, IN1, TAGS, TAGS);
}
